// Attention_5935644803222
// MI455X (gfx1250) — matrix operand order repaired and checked by reading, not compiled, not run on hardware
//
#include <hip/hip_runtime.h>
#include <hip/hip_bf16.h>

#define B_  2
#define S_  2048
#define D_  1024
#define H_  16
#define DH_ 64

typedef __attribute__((ext_vector_type(16))) __bf16 v16bf;
typedef __attribute__((ext_vector_type(8)))  float  v8f;

union FragU { v16bf v; uint4 u[2]; };

__device__ __forceinline__ v8f wmma_bf16(v16bf a, v16bf b, v8f c) {
  // (neg_a, A, neg_b, B, c_mod, C, reuse_a, reuse_b)
  return __builtin_amdgcn_wmma_f32_16x16x32_bf16(false, a, false, b, (short)0, c,
                                                 false, false);
}

// A fragment 16x32 (MxK), row-major source, ld = row stride in elements.
// Lane l: row = l%16; lanes 0-15 hold K {0..7, 16..23}, lanes 16-31 hold K {8..15, 24..31}.
__device__ __forceinline__ v16bf load_frag_a(const __hip_bfloat16* p, int ld, int lane) {
  const int row = lane & 15;
  const int c0  = (lane < 16) ? 0 : 8;
  FragU f;
  const __hip_bfloat16* q = p + (size_t)row * ld + c0;
  f.u[0] = *(const uint4*)(q);
  f.u[1] = *(const uint4*)(q + 16);
  return f.v;
}

// B fragment 32x16 (KxN). Source stores each column as 32 contiguous K elements
// ("B^T row-major"), ld = stride between columns. Lane l: col = l%16,
// lanes 0-15 hold K {0..7, 16..23}, lanes 16-31 hold K {8..15, 24..31}.
__device__ __forceinline__ v16bf load_frag_b(const __hip_bfloat16* p, size_t ld, int lane) {
  const int col = lane & 15;
  const int k0  = (lane < 16) ? 0 : 8;
  FragU f;
  const __hip_bfloat16* q = p + (size_t)col * ld + k0;
  f.u[0] = *(const uint4*)(q);
  f.u[1] = *(const uint4*)(q + 16);
  return f.v;
}

__global__ void cvt_bf16_kernel(const float* __restrict__ in,
                                __hip_bfloat16* __restrict__ out, int n) {
  for (int i = blockIdx.x * blockDim.x + threadIdx.x; i < n;
       i += gridDim.x * blockDim.x)
    out[i] = __float2bfloat16(in[i]);
}

// One wave computes a 16-token x 64-col (one full head) tile of X @ W^T,
// normalizes Q/K rows in-register, and writes bf16 Qn/Kn [B,H,S,DH], Vt [B,H,DH,S].
// Ping-pong double-buffered k-loop with peeled tail: steady-state body has an
// unconditional refill -> single-predecessor latch -> partial s_wait_loadcnt.
__global__ void qkv_proj_kernel(const __hip_bfloat16* __restrict__ Xbf,
                                const __hip_bfloat16* __restrict__ Wqkv,
                                __hip_bfloat16* __restrict__ Qn,
                                __hip_bfloat16* __restrict__ Kn,
                                __hip_bfloat16* __restrict__ Vt) {
  const int lane = threadIdx.x & 31;
  const int wave = threadIdx.x >> 5;
  const int task = blockIdx.x * 4 + wave;          // 0 .. 3*256*16-1
  const int mat  = task / (256 * 16);              // 0=Q 1=K 2=V
  const int rem  = task % (256 * 16);
  const int rowtile = rem >> 4;                    // 0..255
  const int head    = rem & 15;
  const int tbase = rowtile * 16;
  const int nbase = head * DH_;
  const __hip_bfloat16* W  = Wqkv + (size_t)mat * D_ * D_;
  const __hip_bfloat16* Ap = Xbf + (size_t)tbase * D_;
  const __hip_bfloat16* B0 = W + (size_t)(nbase +  0) * D_;
  const __hip_bfloat16* B1 = W + (size_t)(nbase + 16) * D_;
  const __hip_bfloat16* B2 = W + (size_t)(nbase + 32) * D_;
  const __hip_bfloat16* B3 = W + (size_t)(nbase + 48) * D_;

  v8f c0 = {0,0,0,0,0,0,0,0}, c1 = c0, c2 = c0, c3 = c0;

  // set 0: fragments for kb = 0
  v16bf a0  = load_frag_a(Ap, D_, lane);
  v16bf b00 = load_frag_b(B0, D_, lane);
  v16bf b01 = load_frag_b(B1, D_, lane);
  v16bf b02 = load_frag_b(B2, D_, lane);
  v16bf b03 = load_frag_b(B3, D_, lane);

#pragma unroll 1
  for (int kb = 0; kb < D_ - 64; kb += 64) {
    const int k1 = kb + 32;
    __builtin_prefetch(Ap + k1 + 96, 0, 1);
    const v16bf a1  = load_frag_a(Ap + k1, D_, lane);
    const v16bf b10 = load_frag_b(B0 + k1, D_, lane);
    const v16bf b11 = load_frag_b(B1 + k1, D_, lane);
    const v16bf b12 = load_frag_b(B2 + k1, D_, lane);
    const v16bf b13 = load_frag_b(B3 + k1, D_, lane);

    c0 = wmma_bf16(a0, b00, c0);
    c1 = wmma_bf16(a0, b01, c1);
    c2 = wmma_bf16(a0, b02, c2);
    c3 = wmma_bf16(a0, b03, c3);

    const int k2 = kb + 64;                 // unconditional refill (k2 <= D_-64)
    a0  = load_frag_a(Ap + k2, D_, lane);
    b00 = load_frag_b(B0 + k2, D_, lane);
    b01 = load_frag_b(B1 + k2, D_, lane);
    b02 = load_frag_b(B2 + k2, D_, lane);
    b03 = load_frag_b(B3 + k2, D_, lane);

    c0 = wmma_bf16(a1, b10, c0);
    c1 = wmma_bf16(a1, b11, c1);
    c2 = wmma_bf16(a1, b12, c2);
    c3 = wmma_bf16(a1, b13, c3);
  }

  { // peeled tail: set 0 holds kb = D_-64; load kb = D_-32, no refill
    const int k1 = D_ - 32;
    const v16bf a1  = load_frag_a(Ap + k1, D_, lane);
    const v16bf b10 = load_frag_b(B0 + k1, D_, lane);
    const v16bf b11 = load_frag_b(B1 + k1, D_, lane);
    const v16bf b12 = load_frag_b(B2 + k1, D_, lane);
    const v16bf b13 = load_frag_b(B3 + k1, D_, lane);

    c0 = wmma_bf16(a0, b00, c0);
    c1 = wmma_bf16(a0, b01, c1);
    c2 = wmma_bf16(a0, b02, c2);
    c3 = wmma_bf16(a0, b03, c3);
    c0 = wmma_bf16(a1, b10, c0);
    c1 = wmma_bf16(a1, b11, c1);
    c2 = wmma_bf16(a1, b12, c2);
    c3 = wmma_bf16(a1, b13, c3);
  }

  float acc[4][8];
#pragma unroll
  for (int v = 0; v < 8; v++) {
    acc[0][v] = c0[v]; acc[1][v] = c1[v]; acc[2][v] = c2[v]; acc[3][v] = c3[v];
  }

  if (mat < 2) {
    // C-layout: VGPR v, half-group g = lane/16 holds row m = v + 8g, col = lane%16.
    // Row sum-of-squares = reduce over the 16 lanes of the half-group.
#pragma unroll
    for (int v = 0; v < 8; v++) {
      float t = acc[0][v]*acc[0][v] + acc[1][v]*acc[1][v]
              + acc[2][v]*acc[2][v] + acc[3][v]*acc[3][v];
      t += __shfl_xor(t, 1, 32);
      t += __shfl_xor(t, 2, 32);
      t += __shfl_xor(t, 4, 32);
      t += __shfl_xor(t, 8, 32);
      const float r = rsqrtf(t);
      acc[0][v] *= r; acc[1][v] *= r; acc[2][v] *= r; acc[3][v] *= r;
    }
  }

  const int n_lo  = lane & 15;
  const int m_off = (lane >> 4) << 3;
#pragma unroll
  for (int v = 0; v < 8; v++) {
    const int m  = v + m_off;
    const int t  = tbase + m;
    const int bb = t >> 11;           // / S_
    const int s  = t & (S_ - 1);
#pragma unroll
    for (int d = 0; d < 4; d++) {
      const int n = d * 16 + n_lo;    // dh within head
      const __hip_bfloat16 val = __float2bfloat16(acc[d][v]);
      if (mat == 0)
        Qn[(((size_t)bb * H_ + head) * S_ + s) * DH_ + n] = val;
      else if (mat == 1)
        Kn[(((size_t)bb * H_ + head) * S_ + s) * DH_ + n] = val;
      else
        Vt[(((size_t)bb * H_ + head) * DH_ + n) * S_ + s] = val;
    }
  }
}

// Flash-style fused attention: scores -> relu -> causal+pad mask -> @V,
// never materializing the SxS score matrix in global memory.
// V fragments + key-mask words are loaded at the top of each iteration so
// their latency hides behind the score WMMAs and the LDS layout shuffle.
__global__ void attn_kernel(const __hip_bfloat16* __restrict__ Qn,
                            const __hip_bfloat16* __restrict__ Kn,
                            const __hip_bfloat16* __restrict__ Vt,
                            const int* __restrict__ masks,
                            __hip_bfloat16* __restrict__ ctx) {
  __shared__ __align__(16) __hip_bfloat16 lds[8][16][40];  // 40-elem rows: 80B, conflict-free
  const int lane = threadIdx.x & 31;
  const int wave = threadIdx.x >> 5;
  const int qblk = blockIdx.x & 15;      // S_/128 = 16 query blocks
  const int bh   = blockIdx.x >> 4;      // b*H + h
  const int b    = bh >> 4;
  const int h    = bh & 15;
  const int q0   = qblk * 128 + wave * 16;

  const __hip_bfloat16* Qh = Qn + (size_t)bh * S_ * DH_;
  const __hip_bfloat16* Kh = Kn + (size_t)bh * S_ * DH_;
  const __hip_bfloat16* Vh = Vt + (size_t)bh * DH_ * S_;
  const int* mrow = masks + b * S_;

  const v16bf aq0 = load_frag_a(Qh + (size_t)q0 * DH_,      DH_, lane);
  const v16bf aq1 = load_frag_a(Qh + (size_t)q0 * DH_ + 32, DH_, lane);

  v8f acc0 = {0,0,0,0,0,0,0,0}, acc1 = acc0, acc2 = acc0, acc3 = acc0;
  const int n_lo  = lane & 15;
  const int m_off = (lane >> 4) << 3;
  const int kb_end = qblk * 128 + 128;   // uniform across the workgroup -> barriers legal

  for (int kb = 0; kb < kb_end; kb += 32) {
    // Independent loads first: K fragments, V fragments, key masks.
    const v16bf bka0 = load_frag_b(Kh + (size_t)kb * DH_,             DH_, lane);
    const v16bf bkb0 = load_frag_b(Kh + (size_t)kb * DH_ + 32,        DH_, lane);
    const v16bf bka1 = load_frag_b(Kh + (size_t)(kb + 16) * DH_,      DH_, lane);
    const v16bf bkb1 = load_frag_b(Kh + (size_t)(kb + 16) * DH_ + 32, DH_, lane);
    const v16bf bv0  = load_frag_b(Vh + (size_t) 0 * S_ + kb, S_, lane);
    const v16bf bv1  = load_frag_b(Vh + (size_t)16 * S_ + kb, S_, lane);
    const v16bf bv2  = load_frag_b(Vh + (size_t)32 * S_ + kb, S_, lane);
    const v16bf bv3  = load_frag_b(Vh + (size_t)48 * S_ + kb, S_, lane);
    const int  k0i = kb + n_lo;
    const int  k1i = k0i + 16;
    const bool km0 = (mrow[k0i] != 0);
    const bool km1 = (mrow[k1i] != 0);

    v8f s0 = {0,0,0,0,0,0,0,0}, s1 = s0;
    s0 = wmma_bf16(aq0, bka0, s0);
    s0 = wmma_bf16(aq1, bkb0, s0);
    s1 = wmma_bf16(aq0, bka1, s1);
    s1 = wmma_bf16(aq1, bkb1, s1);

#pragma unroll
    for (int v = 0; v < 8; v++) {
      const int q = q0 + v + m_off;
      const float x0 = (km0 && (k0i <= q)) ? fmaxf(s0[v], 0.f) : 0.f;
      const float x1 = (km1 && (k1i <= q)) ? fmaxf(s1[v], 0.f) : 0.f;
      lds[wave][v + m_off][n_lo]      = __float2bfloat16(x0);
      lds[wave][v + m_off][n_lo + 16] = __float2bfloat16(x1);
    }
    __syncthreads();
    const v16bf att = load_frag_a(&lds[wave][0][0], 40, lane);
    acc0 = wmma_bf16(att, bv0, acc0);
    acc1 = wmma_bf16(att, bv1, acc1);
    acc2 = wmma_bf16(att, bv2, acc2);
    acc3 = wmma_bf16(att, bv3, acc3);
    __syncthreads();
  }

#pragma unroll
  for (int v = 0; v < 8; v++) {
    const int   q  = q0 + v + m_off;
    const float qm = (mrow[q] != 0) ? 1.f : 0.f;
    __hip_bfloat16* dst = ctx + ((size_t)(b * S_ + q)) * D_ + h * DH_ + n_lo;
    dst[0]  = __float2bfloat16(acc0[v] * qm);
    dst[16] = __float2bfloat16(acc1[v] * qm);
    dst[32] = __float2bfloat16(acc2[v] * qm);
    dst[48] = __float2bfloat16(acc3[v] * qm);
  }
}

__global__ void out_proj_kernel(const __hip_bfloat16* __restrict__ ctxbf,
                                const __hip_bfloat16* __restrict__ Wobf,
                                float* __restrict__ out) {
  const int lane = threadIdx.x & 31;
  const int wave = threadIdx.x >> 5;
  const int task = blockIdx.x * 4 + wave;   // 0..4095
  const int rowtile = task >> 4;
  const int ncol    = task & 15;
  const int tbase = rowtile * 16;
  const int nbase = ncol * 64;
  const __hip_bfloat16* Ap = ctxbf + (size_t)tbase * D_;
  const __hip_bfloat16* B0 = Wobf + (size_t)(nbase +  0) * D_;
  const __hip_bfloat16* B1 = Wobf + (size_t)(nbase + 16) * D_;
  const __hip_bfloat16* B2 = Wobf + (size_t)(nbase + 32) * D_;
  const __hip_bfloat16* B3 = Wobf + (size_t)(nbase + 48) * D_;

  v8f c0 = {0,0,0,0,0,0,0,0}, c1 = c0, c2 = c0, c3 = c0;

  v16bf a0  = load_frag_a(Ap, D_, lane);
  v16bf b00 = load_frag_b(B0, D_, lane);
  v16bf b01 = load_frag_b(B1, D_, lane);
  v16bf b02 = load_frag_b(B2, D_, lane);
  v16bf b03 = load_frag_b(B3, D_, lane);

#pragma unroll 1
  for (int kb = 0; kb < D_ - 64; kb += 64) {
    const int k1 = kb + 32;
    __builtin_prefetch(Ap + k1 + 96, 0, 1);
    const v16bf a1  = load_frag_a(Ap + k1, D_, lane);
    const v16bf b10 = load_frag_b(B0 + k1, D_, lane);
    const v16bf b11 = load_frag_b(B1 + k1, D_, lane);
    const v16bf b12 = load_frag_b(B2 + k1, D_, lane);
    const v16bf b13 = load_frag_b(B3 + k1, D_, lane);

    c0 = wmma_bf16(a0, b00, c0);
    c1 = wmma_bf16(a0, b01, c1);
    c2 = wmma_bf16(a0, b02, c2);
    c3 = wmma_bf16(a0, b03, c3);

    const int k2 = kb + 64;                 // unconditional refill
    a0  = load_frag_a(Ap + k2, D_, lane);
    b00 = load_frag_b(B0 + k2, D_, lane);
    b01 = load_frag_b(B1 + k2, D_, lane);
    b02 = load_frag_b(B2 + k2, D_, lane);
    b03 = load_frag_b(B3 + k2, D_, lane);

    c0 = wmma_bf16(a1, b10, c0);
    c1 = wmma_bf16(a1, b11, c1);
    c2 = wmma_bf16(a1, b12, c2);
    c3 = wmma_bf16(a1, b13, c3);
  }

  { // peeled tail
    const int k1 = D_ - 32;
    const v16bf a1  = load_frag_a(Ap + k1, D_, lane);
    const v16bf b10 = load_frag_b(B0 + k1, D_, lane);
    const v16bf b11 = load_frag_b(B1 + k1, D_, lane);
    const v16bf b12 = load_frag_b(B2 + k1, D_, lane);
    const v16bf b13 = load_frag_b(B3 + k1, D_, lane);

    c0 = wmma_bf16(a0, b00, c0);
    c1 = wmma_bf16(a0, b01, c1);
    c2 = wmma_bf16(a0, b02, c2);
    c3 = wmma_bf16(a0, b03, c3);
    c0 = wmma_bf16(a1, b10, c0);
    c1 = wmma_bf16(a1, b11, c1);
    c2 = wmma_bf16(a1, b12, c2);
    c3 = wmma_bf16(a1, b13, c3);
  }

  const int n_lo  = lane & 15;
  const int m_off = (lane >> 4) << 3;
#pragma unroll
  for (int v = 0; v < 8; v++) {
    float* dst = out + (size_t)(tbase + v + m_off) * D_ + nbase + n_lo;
    dst[0]  = c0[v];
    dst[16] = c1[v];
    dst[32] = c2[v];
    dst[48] = c3[v];
  }
}

extern "C" void kernel_launch(void* const* d_in, const int* in_sizes, int n_in,
                              void* d_out, int out_size, void* d_ws, size_t ws_size,
                              hipStream_t stream) {
  const float* X     = (const float*)d_in[0];
  const int*   masks = (const int*)d_in[1];
  const float* Wq    = (const float*)d_in[2];
  const float* Wk    = (const float*)d_in[3];
  const float* Wv    = (const float*)d_in[4];
  const float* Wo    = (const float*)d_in[5];
  float* out = (float*)d_out;

  const size_t ND = (size_t)B_ * S_ * D_;   // 4,194,304
  const size_t DD = (size_t)D_ * D_;        // 1,048,576

  __hip_bfloat16* p    = (__hip_bfloat16*)d_ws;
  __hip_bfloat16* Xbf  = p;  p += ND;
  __hip_bfloat16* Wqkv = p;  p += 3 * DD;
  __hip_bfloat16* Wobf = p;  p += DD;
  __hip_bfloat16* Qn   = p;  p += ND;
  __hip_bfloat16* Kn   = p;  p += ND;
  __hip_bfloat16* Vt   = p;  p += ND;
  __hip_bfloat16* Ctx  = p;  p += ND;
  (void)ws_size; (void)in_sizes; (void)n_in; (void)out_size;

  cvt_bf16_kernel<<<1024, 256, 0, stream>>>(X,  Xbf,          (int)ND);
  cvt_bf16_kernel<<<256,  256, 0, stream>>>(Wq, Wqkv + 0 * DD, (int)DD);
  cvt_bf16_kernel<<<256,  256, 0, stream>>>(Wk, Wqkv + 1 * DD, (int)DD);
  cvt_bf16_kernel<<<256,  256, 0, stream>>>(Wv, Wqkv + 2 * DD, (int)DD);
  cvt_bf16_kernel<<<256,  256, 0, stream>>>(Wo, Wobf,          (int)DD);

  // 3 mats * 256 rowtiles * 16 heads = 12288 wave tasks, 4 waves/block
  qkv_proj_kernel<<<3072, 128, 0, stream>>>(Xbf, Wqkv, Qn, Kn, Vt);
  // B*H*(S/128) = 512 workgroups of 8 waves
  attn_kernel<<<512, 256, 0, stream>>>(Qn, Kn, Vt, masks, Ctx);
  // 256 rowtiles * 16 col tiles = 4096 wave tasks
  out_proj_kernel<<<1024, 128, 0, stream>>>(Ctx, Wobf, out);
}
